// TimeAwareMHA_33131377721445
// MI455X (gfx1250) — matrix operand order repaired and checked by reading, not compiled, not run on hardware
//
#include <hip/hip_runtime.h>

// ---------------------------------------------------------------------------
// Time-aware MHA for MI455X (gfx1250, wave32, WMMA bf16 16x16x32)
// Double-buffered LDS pipelines, b128 staging, fused epilogues.
// ---------------------------------------------------------------------------
#define D_MODEL   1024
#define N_HEADS   16
#define DK        64
#define BATCH     8
#define SEQ_LEN   1024
#define ROWS      (BATCH * SEQ_LEN)   // 8192
#define NEG_INF   (-1.0e9f)

typedef __attribute__((ext_vector_type(16))) __bf16 bf16x16;
typedef __attribute__((ext_vector_type(8)))  __bf16 bf16x8;
typedef __attribute__((ext_vector_type(8)))  float  f32x8;

static __device__ __forceinline__ f32x8 f32x8_zero() {
    f32x8 z;
#pragma unroll
    for (int i = 0; i < 8; ++i) z[i] = 0.0f;
    return z;
}

// Combine two 16-byte loads into one 16-element bf16 fragment.
static __device__ __forceinline__ bf16x16 frag2(const __bf16* p0, const __bf16* p1) {
    bf16x8 lo = *(const bf16x8*)p0;
    bf16x8 hi = *(const bf16x8*)p1;
    return __builtin_shufflevector(lo, hi, 0,1,2,3,4,5,6,7,8,9,10,11,12,13,14,15);
}

static __device__ __forceinline__ f32x8 wmma_bf16(bf16x16 a, bf16x16 b, f32x8 c) {
    // (neg_a, A, neg_b, B, c_mod, C, reuse_a, reuse_b)
    return __builtin_amdgcn_wmma_f32_16x16x32_bf16(false, a, false, b, (short)0, c,
                                                   false, false);
}

// ---------------------------------------------------------------------------
// elementwise f32 -> bf16
// ---------------------------------------------------------------------------
__global__ __launch_bounds__(256) void cast_f32_bf16(const float* __restrict__ in,
                                                     __bf16* __restrict__ out, int n) {
    int i = blockIdx.x * 256 + threadIdx.x;
    if (i < n) out[i] = (__bf16)in[i];
}

// W (N x K, row-major) -> Wt (K x N, row-major) bf16
__global__ __launch_bounds__(256) void transpose_cast(const float* __restrict__ W,
                                                      __bf16* __restrict__ Wt) {
    int i = blockIdx.x * 256 + threadIdx.x;          // 0 .. D*D-1
    int k = i >> 10;
    int n = i & (D_MODEL - 1);
    Wt[i] = (__bf16)W[n * D_MODEL + k];
}

// ---------------------------------------------------------------------------
// GEMM:  out[M x D] = A[M x D](bf16) @ Bt[D x D](bf16, K-major)  + epilogue
// block tile 128x64, 8 waves (4m x 2n), each wave 32x32 via 4 WMMA accum tiles
// double-buffered LDS, one barrier per K-step, global loads overlap WMMA
// ---------------------------------------------------------------------------
#define AST 40   // LDS stride (elems) for A tile rows (128 x 32)
#define BST 72   // LDS stride (elems) for B tile rows (32 x 64)
#define KSTEPS (D_MODEL / 32)

__global__ __launch_bounds__(256) void gemm_bf16(
    const __bf16* __restrict__ A, const __bf16* __restrict__ Bt,
    const float* __restrict__ bias,
    const float* __restrict__ pos,      // (L x D) or null
    const float* __restrict__ intTab,   // (TB+1 x D) or null
    const int*   __restrict__ timemat,  // (B x L x L) or null
    __bf16* __restrict__ outBf, float* __restrict__ outF)
{
    __shared__ __bf16 ldsA[2][128 * AST];
    __shared__ __bf16 ldsB[2][32 * BST];

    const int tid  = threadIdx.x;
    const int wid  = tid >> 5;
    const int lane = tid & 31;
    const int hgrp = lane >> 4;      // half-group (0: lanes 0-15, 1: 16-31)
    const int ln16 = lane & 15;
    const int wm   = wid & 3;        // wave m index (0..3)
    const int wn   = wid >> 2;       // wave n index (0..1)
    const int m0   = blockIdx.x * 128;
    const int n0   = blockIdx.y * 64;

    // staging indices: A 128x32 as 512 b128 loads (2 per thread), B 32x64 as 256
    const int aRow0 = tid >> 2,         aCg0 = tid & 3;
    const int aRow1 = (tid + 256) >> 2, aCg1 = (tid + 256) & 3;
    const int bRow  = tid >> 3,         bCg  = tid & 7;

    f32x8 acc[2][2];
#pragma unroll
    for (int mi = 0; mi < 2; ++mi)
#pragma unroll
        for (int ni = 0; ni < 2; ++ni) acc[mi][ni] = f32x8_zero();

    // preload tile 0 into registers
    bf16x8 ra0 = *(const bf16x8*)&A[(size_t)(m0 + aRow0) * D_MODEL + aCg0 * 8];
    bf16x8 ra1 = *(const bf16x8*)&A[(size_t)(m0 + aRow1) * D_MODEL + aCg1 * 8];
    bf16x8 rb  = *(const bf16x8*)&Bt[(size_t)bRow * D_MODEL + n0 + bCg * 8];

    for (int kt = 0; kt < KSTEPS; ++kt) {
        const int cur = kt & 1;
        // commit staged registers to the current LDS buffer
        *(bf16x8*)&ldsA[cur][aRow0 * AST + aCg0 * 8] = ra0;
        *(bf16x8*)&ldsA[cur][aRow1 * AST + aCg1 * 8] = ra1;
        *(bf16x8*)&ldsB[cur][bRow * BST + bCg * 8]   = rb;
        __syncthreads();

        // overlap: fetch next tile into registers while computing this one
        if (kt + 1 < KSTEPS) {
            const int k1 = (kt + 1) * 32;
            ra0 = *(const bf16x8*)&A[(size_t)(m0 + aRow0) * D_MODEL + k1 + aCg0 * 8];
            ra1 = *(const bf16x8*)&A[(size_t)(m0 + aRow1) * D_MODEL + k1 + aCg1 * 8];
            rb  = *(const bf16x8*)&Bt[(size_t)(k1 + bRow) * D_MODEL + n0 + bCg * 8];
            if (kt + 2 < KSTEPS)
                __builtin_prefetch(&A[(size_t)(m0 + aRow0) * D_MODEL + k1 + 32], 0, 0);
        }

        bf16x16 af[2], bfr[2];
#pragma unroll
        for (int mi = 0; mi < 2; ++mi) {
            // A frag 16x32: lane row = wm*32 + mi*16 + ln16
            // K(e) = (e/8)*16 + hgrp*8 + e%8  -> two 16B runs at +hgrp*8, +16+hgrp*8
            const __bf16* rp = &ldsA[cur][(wm * 32 + mi * 16 + ln16) * AST + hgrp * 8];
            af[mi] = frag2(rp, rp + 16);
        }
#pragma unroll
        for (int ni = 0; ni < 2; ++ni) {
            // B frag 32x16: lane col n = ln16, K(e) = (e/8)*16 + hgrp*8 + e%8
#pragma unroll
            for (int e = 0; e < 16; ++e)
                bfr[ni][e] = ldsB[cur][((e >> 3) * 16 + hgrp * 8 + (e & 7)) * BST + wn * 32 + ni * 16 + ln16];
        }
#pragma unroll
        for (int mi = 0; mi < 2; ++mi)
#pragma unroll
            for (int ni = 0; ni < 2; ++ni)
                acc[mi][ni] = wmma_bf16(af[mi], bfr[ni], acc[mi][ni]);
    }

    // epilogue: C layout lane/reg -> (m = r + 8*hgrp, n = ln16)
#pragma unroll
    for (int mi = 0; mi < 2; ++mi) {
#pragma unroll
        for (int ni = 0; ni < 2; ++ni) {
            int col = n0 + wn * 32 + ni * 16 + ln16;
            float bv = bias[col];
#pragma unroll
            for (int r = 0; r < 8; ++r) {
                int grow = m0 + wm * 32 + mi * 16 + r + 8 * hgrp;
                float v = acc[mi][ni][r] + bv;
                if (pos) {
                    int li = grow & (SEQ_LEN - 1);
                    v += pos[li * D_MODEL + col];
                    if (timemat) {
                        int bi = grow >> 10;
                        int t = timemat[(size_t)(bi * SEQ_LEN + li) * SEQ_LEN];
                        v += intTab[(size_t)t * D_MODEL + col];
                    }
                }
                size_t oidx = (size_t)grow * D_MODEL + col;
                if (outBf) outBf[oidx] = (__bf16)v;
                else       outF[oidx]  = v;
            }
        }
    }
}

// ---------------------------------------------------------------------------
// Flash attention: block = (b, h, 128 q rows), 8 waves x 16 q rows each.
// Key loop in 32-key tiles, double-buffered K/V staging, causal early skip.
// ---------------------------------------------------------------------------
#define KTST 40  // LDS K-transposed (64 x 32) row stride
#define VST  72  // LDS V (32 x 64) row stride
#define PST  40  // per-wave P scratch (16 x 32) row stride

__global__ __launch_bounds__(256) void flash_attn(
    const __bf16* __restrict__ Q, const __bf16* __restrict__ K,
    const __bf16* __restrict__ V, const unsigned char* __restrict__ pad,
    __bf16* __restrict__ ctx)
{
    __shared__ __bf16 ldsKT[2][64 * KTST];       // [c][key]
    __shared__ __bf16 ldsV [2][32 * VST];        // [key][c]
    __shared__ __bf16 ldsP [8 * 16 * PST];       // per-wave 16x32

    const int qt = blockIdx.x;   // q tile (0..7)
    const int h  = blockIdx.y;
    const int b  = blockIdx.z;
    const int tid  = threadIdx.x;
    const int wid  = tid >> 5;
    const int lane = tid & 31;
    const int hgrp = lane >> 4;
    const int ln16 = lane & 15;
    const int q0   = qt * 128;
    const int qb   = q0 + wid * 16;           // this wave's first q row
    __bf16* myP = &ldsP[wid * 16 * PST];

    // Q fragments for this wave's 16 rows (dk=64 -> two 16x32 A-frags)
    const __bf16* qrow = Q + ((size_t)(b * SEQ_LEN + qb + ln16)) * D_MODEL + h * DK;
    bf16x16 qf[2];
#pragma unroll
    for (int j = 0; j < 2; ++j) {
        const __bf16* p = qrow + j * 32 + hgrp * 8;
        qf[j] = frag2(p, p + 16);
    }

    float m_i[8], l_i[8];
    f32x8 o[4];
#pragma unroll
    for (int r = 0; r < 8; ++r) { m_i[r] = -3.0e38f; l_i[r] = 0.0f; }
#pragma unroll
    for (int ct = 0; ct < 4; ++ct) o[ct] = f32x8_zero();

    const int nTiles = (q0 + 128) / 32;       // causal bound for the block
    const int key = tid >> 3, cg = tid & 7;   // staging coords: 32 keys x 8 groups
    const __bf16* kBase = K + ((size_t)(b * SEQ_LEN + key)) * D_MODEL + h * DK + cg * 8;
    const __bf16* vBase = V + ((size_t)(b * SEQ_LEN + key)) * D_MODEL + h * DK + cg * 8;

    // preload tile 0
    bf16x8 rK = *(const bf16x8*)kBase;
    bf16x8 rV = *(const bf16x8*)vBase;

    for (int t = 0; t < nTiles; ++t) {
        const int cur = t & 1;
        const int kb  = t * 32;
        // commit staged K (transposed scatter) and V (b128) to LDS
#pragma unroll
        for (int i = 0; i < 8; ++i) ldsKT[cur][(cg * 8 + i) * KTST + key] = rK[i];
        *(bf16x8*)&ldsV[cur][key * VST + cg * 8] = rV;
        __syncthreads();

        // overlap next tile's global loads with compute
        if (t + 1 < nTiles) {
            rK = *(const bf16x8*)(kBase + (size_t)(kb + 32) * D_MODEL);
            rV = *(const bf16x8*)(vBase + (size_t)(kb + 32) * D_MODEL);
            if (t + 2 < nTiles)
                __builtin_prefetch(kBase + (size_t)(kb + 64) * D_MODEL, 0, 0);
        }

        if (kb <= qb + 15) {                  // wave has unmasked keys in tile
            // S = Q K'^T for two 16-key subtiles
            f32x8 s[2];
#pragma unroll
            for (int si = 0; si < 2; ++si) {
                f32x8 accS = f32x8_zero();
#pragma unroll
                for (int j = 0; j < 2; ++j) {
                    bf16x16 kf;
#pragma unroll
                    for (int e = 0; e < 16; ++e)
                        kf[e] = ldsKT[cur][(j * 32 + (e >> 3) * 16 + hgrp * 8 + (e & 7)) * KTST + si * 16 + ln16];
                    accS = wmma_bf16(qf[j], kf, accS);
                }
                s[si] = accS;
            }
            // scale + causal/pad mask (element (r): m = r+8*hgrp, n = ln16)
#pragma unroll
            for (int si = 0; si < 2; ++si) {
                int kidx = kb + si * 16 + ln16;
                bool pk = pad[b * SEQ_LEN + kidx] != 0;
#pragma unroll
                for (int r = 0; r < 8; ++r) {
                    int qr = qb + r + 8 * hgrp;
                    float v = s[si][r] * 0.125f;
                    s[si][r] = (kidx > qr || pk) ? NEG_INF : v;
                }
            }
            // online softmax (row stats shared across 16-lane half-group)
#pragma unroll
            for (int r = 0; r < 8; ++r) {
                float tmax = fmaxf(s[0][r], s[1][r]);
#pragma unroll
                for (int msk = 8; msk >= 1; msk >>= 1)
                    tmax = fmaxf(tmax, __shfl_xor(tmax, msk, 32));
                float mnew  = fmaxf(m_i[r], tmax);
                float alpha = __expf(m_i[r] - mnew);
                float p0 = __expf(s[0][r] - mnew);
                float p1 = __expf(s[1][r] - mnew);
                float rs = p0 + p1;
#pragma unroll
                for (int msk = 8; msk >= 1; msk >>= 1)
                    rs += __shfl_xor(rs, msk, 32);
                l_i[r] = l_i[r] * alpha + rs;
                m_i[r] = mnew;
#pragma unroll
                for (int ct = 0; ct < 4; ++ct) o[ct][r] *= alpha;
                __bf16* prow = &myP[(r + 8 * hgrp) * PST];
                prow[ln16]      = (__bf16)p0;
                prow[16 + ln16] = (__bf16)p1;
            }
            // O += P @ V   (A-frag from wave-local P scratch; same-wave LDS RAW,
            // DS pipe is in-order per wave)
            const __bf16* pp = &myP[ln16 * PST + hgrp * 8];
            bf16x16 pf = frag2(pp, pp + 16);
#pragma unroll
            for (int ct = 0; ct < 4; ++ct) {
                bf16x16 vf;
#pragma unroll
                for (int e = 0; e < 16; ++e)
                    vf[e] = ldsV[cur][((e >> 3) * 16 + hgrp * 8 + (e & 7)) * VST + ct * 16 + ln16];
                o[ct] = wmma_bf16(pf, vf, o[ct]);
            }
        }
    }

    // normalize + write ctx (bf16)
#pragma unroll
    for (int ct = 0; ct < 4; ++ct) {
        int col = h * DK + ct * 16 + ln16;
#pragma unroll
        for (int r = 0; r < 8; ++r) {
            int qr = qb + r + 8 * hgrp;
            float v = o[ct][r] / l_i[r];
            ctx[((size_t)(b * SEQ_LEN + qr)) * D_MODEL + col] = (__bf16)v;
        }
    }
}

// ---------------------------------------------------------------------------
// host-side launch
// ---------------------------------------------------------------------------
extern "C" void kernel_launch(void* const* d_in, const int* in_sizes, int n_in,
                              void* d_out, int out_size, void* d_ws, size_t ws_size,
                              hipStream_t stream) {
    (void)in_sizes; (void)n_in; (void)out_size; (void)ws_size;
    const float* seq      = (const float*)d_in[0];
    const int*   timemat  = (const int*)d_in[1];
    const unsigned char* pad = (const unsigned char*)d_in[2];
    const float* Wq = (const float*)d_in[3];  const float* bq = (const float*)d_in[4];
    const float* Wk = (const float*)d_in[5];  const float* bk = (const float*)d_in[6];
    const float* Wv = (const float*)d_in[7];  const float* bv = (const float*)d_in[8];
    const float* Wo = (const float*)d_in[9];  const float* bo = (const float*)d_in[10];
    const float* absK = (const float*)d_in[11];
    const float* absV = (const float*)d_in[12];
    const float* intK = (const float*)d_in[13];
    const float* intV = (const float*)d_in[14];

    char* ws = (char*)d_ws;
    size_t off = 0;
    __bf16* seq_bf = (__bf16*)(ws + off); off += (size_t)ROWS * D_MODEL * 2;   // 16MB
    __bf16* WqT = (__bf16*)(ws + off); off += (size_t)D_MODEL * D_MODEL * 2;
    __bf16* WkT = (__bf16*)(ws + off); off += (size_t)D_MODEL * D_MODEL * 2;
    __bf16* WvT = (__bf16*)(ws + off); off += (size_t)D_MODEL * D_MODEL * 2;
    __bf16* WoT = (__bf16*)(ws + off); off += (size_t)D_MODEL * D_MODEL * 2;
    __bf16* Qb  = (__bf16*)(ws + off); off += (size_t)ROWS * D_MODEL * 2;
    __bf16* Kb  = (__bf16*)(ws + off); off += (size_t)ROWS * D_MODEL * 2;
    __bf16* Vb  = (__bf16*)(ws + off); off += (size_t)ROWS * D_MODEL * 2;
    __bf16* Cb  = (__bf16*)(ws + off); off += (size_t)ROWS * D_MODEL * 2;

    const int nSeq = ROWS * D_MODEL;
    cast_f32_bf16<<<nSeq / 256, 256, 0, stream>>>(seq, seq_bf, nSeq);
    const int nW = D_MODEL * D_MODEL;
    transpose_cast<<<nW / 256, 256, 0, stream>>>(Wq, WqT);
    transpose_cast<<<nW / 256, 256, 0, stream>>>(Wk, WkT);
    transpose_cast<<<nW / 256, 256, 0, stream>>>(Wv, WvT);
    transpose_cast<<<nW / 256, 256, 0, stream>>>(Wo, WoT);

    dim3 gGemm(ROWS / 128, D_MODEL / 64);
    // Q = seq @ Wq^T + bq
    gemm_bf16<<<gGemm, 256, 0, stream>>>(seq_bf, WqT, bq, nullptr, nullptr, nullptr,
                                         Qb, nullptr);
    // K' = seq @ Wk^T + bk + absK[l] + intK[timemat[:,:,0]]
    gemm_bf16<<<gGemm, 256, 0, stream>>>(seq_bf, WkT, bk, absK, intK, timemat,
                                         Kb, nullptr);
    // V' = seq @ Wv^T + bv + absV[l] + intV[timemat[:,:,0]]
    gemm_bf16<<<gGemm, 256, 0, stream>>>(seq_bf, WvT, bv, absV, intV, timemat,
                                         Vb, nullptr);

    dim3 gAttn(SEQ_LEN / 128, N_HEADS, BATCH);
    flash_attn<<<gAttn, 256, 0, stream>>>(Qb, Kb, Vb, pad, Cb);

    // out = ctx @ Wo^T + bo  (f32 output)
    gemm_bf16<<<gGemm, 256, 0, stream>>>(Cb, WoT, bo, nullptr, nullptr, nullptr,
                                         nullptr, (float*)d_out);
}
